// TransformerRLPolicy_91319594647599
// MI455X (gfx1250) — matrix operand order repaired and checked by reading, not compiled, not run on hardware
//
#include <hip/hip_runtime.h>
#include <hip/hip_bf16.h>

// ---------------------------------------------------------------------------
// TransformerRLPolicy on gfx1250 (MI455X).
// Compute-bound (~2.8 TFLOP, ~25MB HBM traffic; weights L2-resident) ->
// all GEMMs via v_wmma_f32_16x16x32_f16. One workgroup (8 wave32) per batch
// sample, 64x256 activations resident in LDS. Weights pre-converted to f16
// transposed N-major so the WMMA B operand is two 16B k-contiguous lane loads.
// Register blocking: 4 N-tiles per wave -> A fragment reused 4x per K-step.
// ---------------------------------------------------------------------------

typedef __attribute__((ext_vector_type(16))) _Float16 v16h;
typedef __attribute__((ext_vector_type(8)))  _Float16 v8h;
typedef __attribute__((ext_vector_type(8)))  float    v8f;

#define DEV __device__ __forceinline__

static constexpr int OBSN  = 204;     // 4 + 4*50
static constexpr int SP    = 64;      // padded token count (S=53)
static constexpr int D     = 256;
static constexpr int F     = 1024;

// ---- dynamic LDS layout (bytes) -------------------------------------------
static constexpr int XF_OFF   = 0;                     // 64x256 f32  (65536)
static constexpr int XH_OFF   = 65536;                 // 64x256 f16  (32768)
static constexpr int MASK_OFF = 98304;                 // 64 f32      (256)
static constexpr int OBS_OFF  = 98560;                 // 208 f32     (832)
static constexpr int SCR_OFF  = 99392;                 // scratch     (147456)
static constexpr int SMEM_BYTES = SCR_OFF + 147456;    // 246848 < 320KB

// ---- WMMA fragment loaders (ISA 7.12.2 layouts, wave32) -------------------
DEV v8h ld8(const _Float16* p) { return *(const v8h*)p; }

DEV v16h cat16(v8h a, v8h b) {
  v16h r;
#pragma unroll
  for (int i = 0; i < 8; ++i) { r[i] = a[i]; r[i + 8] = b[i]; }
  return r;
}

// A: 16x32 f16 from row-major [m][k]; lane m=lane&15, half h=lane>>4
DEV v16h loadA(const _Float16* base, int ld, int m0, int k0) {
  const int lane = threadIdx.x & 31;
  const _Float16* row = base + (m0 + (lane & 15)) * ld + k0 + 8 * (lane >> 4);
  return cat16(ld8(row), ld8(row + 16));
}

// B: 32x16 f16 from transposed Bt[n][k] (k contiguous)
DEV v16h loadB(const _Float16* bt, int ldk, int n0, int k0) {
  const int lane = threadIdx.x & 31;
  const _Float16* p = bt + (n0 + (lane & 15)) * ldk + k0 + 8 * (lane >> 4);
  return cat16(ld8(p), ld8(p + 16));
}

DEV v8f wmma16(v16h a, v16h b, v8f c) {
  return __builtin_amdgcn_wmma_f32_16x16x32_f16(false, a, false, b,
                                                (short)0, c, false, false);
}

// Register-blocked GEMM: one 16-row M tile x four 16-col N tiles, depth KT*32.
// A fragment loaded once per K-step and reused for 4 WMMAs.
template <int KT>
DEV void gemm4(const _Float16* Abase, int lda, int m0,
               const _Float16* Bt, int ldk, int n0, v8f acc[4]) {
#pragma unroll
  for (int kt = 0; kt < KT; ++kt) {
    const v16h a = loadA(Abase, lda, m0, kt * 32);
#pragma unroll
    for (int j = 0; j < 4; ++j)
      acc[j] = wmma16(a, loadB(Bt, ldk, n0 + j * 16, kt * 32), acc[j]);
  }
}

// C/D mapping: VGPR r, lanes 0-15 -> M=r, lanes 16-31 -> M=8+r, N=lane&15

// ---- weight prep: f32 [K][N] -> f16 transposed [N][K] ---------------------
__global__ void wprep_kernel(_Float16* __restrict__ dst,
                             const float* __restrict__ src, int K, int N) {
  int idx = blockIdx.x * blockDim.x + threadIdx.x;
  if (idx >= K * N) return;
  int n = idx / K, k = idx - n * K;
  dst[idx] = (_Float16)src[k * N + n];
}

// ---- main per-sample transformer kernel -----------------------------------
struct PolicyParams {
  const float* obs;
  const float* cls_emb; const float* type_cls; const float* type_ctx;
  const float* type_pred; const float* type_boid;
  const float* ctx_W; const float* ctx_b;
  const float* pred_W; const float* pred_b;
  const float* boid_W; const float* boid_b;
  const _Float16* qkvWt; const float* qkv_b;
  const _Float16* attnWt; const float* attn_b;
  const float* ln1_g; const float* ln1_b; const float* ln2_g; const float* ln2_b;
  const _Float16* f1Wt; const float* f1_b;
  const _Float16* f2Wt; const float* f2_b;
  _Float16* clsOut;
};

DEV void layer_norm(float* xf, _Float16* xh, const float* g, const float* bta,
                    int wave, int lane) {
  for (int row = wave; row < SP; row += 8) {
    const float* xr = xf + row * D;
    float s = 0.f, s2 = 0.f, vals[8];
#pragma unroll
    for (int i = 0; i < 8; ++i) {
      float v = xr[lane * 8 + i];
      vals[i] = v; s += v; s2 += v * v;
    }
#pragma unroll
    for (int ms = 1; ms < 32; ms <<= 1) {
      s  += __shfl_xor(s,  ms);
      s2 += __shfl_xor(s2, ms);
    }
    const float mean = s * (1.f / 256.f);
    const float var  = s2 * (1.f / 256.f) - mean * mean;
    const float rs   = rsqrtf(var + 1e-5f);
#pragma unroll
    for (int i = 0; i < 8; ++i) {
      int c = lane * 8 + i;
      float y = (vals[i] - mean) * rs * g[c] + bta[c];
      xf[row * D + c] = y;
      xh[row * D + c] = (_Float16)y;
    }
  }
}

__launch_bounds__(256)
__global__ void policy_kernel(PolicyParams P) {
  extern __shared__ char smem[];
  float*    xf   = (float*)(smem + XF_OFF);
  _Float16* xh   = (_Float16*)(smem + XH_OFF);
  float*    mneg = (float*)(smem + MASK_OFF);
  float*    obsL = (float*)(smem + OBS_OFF);
  _Float16* scr  = (_Float16*)(smem + SCR_OFF);
  _Float16* qbuf = scr;                 // [64][256]
  _Float16* kbuf = scr + 16384;         // [64][256] token-major -> B of Q.K^T
  _Float16* vt   = scr + 32768;         // [256][64] dim-major  -> B of P.V
  _Float16* obuf = scr + 49152;         // [64][256]
  _Float16* pb   = scr + 65536;         // [8 waves][16][64] softmax probs
  _Float16* hbuf = scr;                 // FFN hidden [64][1024] (reuse)

  const int tid  = threadIdx.x;
  const int wave = tid >> 5;
  const int lane = tid & 31;
  const int nl   = lane & 15;
  const int hh   = lane >> 4;
  const int b    = blockIdx.x;

  // ---- load observation + pad mask ----
  const float* orow = P.obs + (size_t)b * OBSN;
  for (int i = tid; i < OBSN; i += 256) obsL[i] = orow[i];
  __syncthreads();
  if (tid < SP) {
    float mn;
    if (tid < 3) mn = 0.f;
    else if (tid < 53) {
      const float* bd = obsL + 4 + (tid - 3) * 4;
      bool pad = fabsf(bd[0]) < 1e-6f && fabsf(bd[1]) < 1e-6f &&
                 fabsf(bd[2]) < 1e-6f && fabsf(bd[3]) < 1e-6f;
      mn = pad ? -1e9f : 0.f;
    } else mn = -1e9f;
    mneg[tid] = mn;
  }
  __syncthreads();

  // ---- token embedding (K<=4: VALU) ----
  for (int idx = tid; idx < SP * D; idx += 256) {
    const int t = idx >> 8, d = idx & 255;
    float v;
    if (t == 0)      v = P.cls_emb[d] + P.type_cls[d];
    else if (t == 1) v = obsL[0] * P.ctx_W[d] + obsL[1] * P.ctx_W[D + d] +
                         P.ctx_b[d] + P.type_ctx[d];
    else if (t == 2) v = obsL[2] * P.pred_W[d] + obsL[3] * P.pred_W[D + d] +
                         P.pred_b[d] + P.type_pred[d];
    else if (t < 53) {
      if (mneg[t] != 0.f) v = 0.f;
      else {
        const float* bd = obsL + 4 + (t - 3) * 4;
        v = bd[0] * P.boid_W[d]       + bd[1] * P.boid_W[D + d] +
            bd[2] * P.boid_W[2*D + d] + bd[3] * P.boid_W[3*D + d] +
            P.boid_b[d] + P.type_boid[d];
      }
    } else v = 0.f;
    xf[idx] = v;
    xh[idx] = (_Float16)v;
  }
  __syncthreads();

  const float scale = 0.17677669529663687f; // 1/sqrt(32)

  for (int l = 0; l < 4; ++l) {
    const _Float16* Wqkv = P.qkvWt  + (size_t)l * 768 * D;
    const _Float16* Wao  = P.attnWt + (size_t)l * D * D;
    const _Float16* W1   = P.f1Wt   + (size_t)l * F * D;
    const _Float16* W2   = P.f2Wt   + (size_t)l * D * F;

    // ---- QKV GEMM: [64x256] @ [256x768], 48 groups of (16m x 64n) ----
    for (int g = wave; g < 48; g += 8) {
      const int mt = g & 3;
      const int n0 = __builtin_amdgcn_readfirstlane((g >> 2) * 64);
      v8f acc[4] = {{}, {}, {}, {}};
      gemm4<8>(xh, D, mt * 16, Wqkv, D, n0, acc);
#pragma unroll
      for (int j = 0; j < 4; ++j) {
        const int col = n0 + j * 16 + nl;
        const float bias = P.qkv_b[l * 768 + col];
        if (n0 < 256) {            // Q (scalar branch: n0 uniform)
#pragma unroll
          for (int r = 0; r < 8; ++r)
            qbuf[(mt * 16 + 8 * hh + r) * D + col] =
                (_Float16)(acc[j][r] + bias);
        } else if (n0 < 512) {     // K
          const int c = col - 256;
#pragma unroll
          for (int r = 0; r < 8; ++r)
            kbuf[(mt * 16 + 8 * hh + r) * D + c] =
                (_Float16)(acc[j][r] + bias);
        } else {                   // V, stored transposed [dim][token]
          const int c = col - 512;
#pragma unroll
          for (int r = 0; r < 8; ++r)
            vt[c * SP + (mt * 16 + 8 * hh + r)] =
                (_Float16)(acc[j][r] + bias);
        }
      }
    }
    __syncthreads();

    // ---- attention: per (head, m-tile); Q fragment reused for 4 K-tiles ----
    for (int it = wave; it < 32; it += 8) {
      const int head = it >> 2, mt = it & 3;
      const v16h aq = loadA(qbuf, D, mt * 16, head * 32);
      v8f s[4];
#pragma unroll
      for (int t = 0; t < 4; ++t) {
        v8f z = {};
        s[t] = wmma16(aq, loadB(kbuf, D, t * 16, head * 32), z);
      }
#pragma unroll
      for (int t = 0; t < 4; ++t) {
        const float mn = mneg[t * 16 + nl];
#pragma unroll
        for (int r = 0; r < 8; ++r) s[t][r] = s[t][r] * scale + mn;
      }
      _Float16* pw = pb + wave * 16 * SP;
#pragma unroll
      for (int r = 0; r < 8; ++r) {
        float mx = fmaxf(fmaxf(s[0][r], s[1][r]), fmaxf(s[2][r], s[3][r]));
#pragma unroll
        for (int ms = 1; ms < 16; ms <<= 1) mx = fmaxf(mx, __shfl_xor(mx, ms));
        float e0 = __expf(s[0][r] - mx), e1 = __expf(s[1][r] - mx);
        float e2 = __expf(s[2][r] - mx), e3 = __expf(s[3][r] - mx);
        float sum = e0 + e1 + e2 + e3;
#pragma unroll
        for (int ms = 1; ms < 16; ms <<= 1) sum += __shfl_xor(sum, ms);
        const float inv = 1.f / sum;
        const int row = 8 * hh + r;
        pw[row * SP +      nl] = (_Float16)(e0 * inv);
        pw[row * SP + 16 + nl] = (_Float16)(e1 * inv);
        pw[row * SP + 32 + nl] = (_Float16)(e2 * inv);
        pw[row * SP + 48 + nl] = (_Float16)(e3 * inv);
      }
      // o = probs[16x64] @ V[64x32]; prob fragments reused across 2 halves
#pragma unroll
      for (int kc = 0; kc < 2; ++kc) {
        const v16h ap = loadA(pw, SP, 0, kc * 32);
        if (kc == 0) {
#pragma unroll
          for (int half = 0; half < 2; ++half) {
            v8f z = {};
            s[half] = wmma16(ap, loadB(vt + head * 32 * SP, SP, half * 16, 0), z);
          }
        } else {
#pragma unroll
          for (int half = 0; half < 2; ++half)
            s[half] = wmma16(ap, loadB(vt + head * 32 * SP, SP, half * 16, 32),
                             s[half]);
        }
      }
#pragma unroll
      for (int half = 0; half < 2; ++half) {
        const int col = head * 32 + half * 16 + nl;
#pragma unroll
        for (int r = 0; r < 8; ++r)
          obuf[(mt * 16 + 8 * hh + r) * D + col] = (_Float16)s[half][r];
      }
    }
    __syncthreads();

    // ---- attn_out proj + residual: 16 groups ----
    for (int g = wave; g < 16; g += 8) {
      const int mt = g & 3;
      const int n0 = __builtin_amdgcn_readfirstlane((g >> 2) * 64);
      v8f acc[4] = {{}, {}, {}, {}};
      gemm4<8>(obuf, D, mt * 16, Wao, D, n0, acc);
#pragma unroll
      for (int j = 0; j < 4; ++j) {
        const int col = n0 + j * 16 + nl;
        const float bias = P.attn_b[l * D + col];
#pragma unroll
        for (int r = 0; r < 8; ++r)
          xf[(mt * 16 + 8 * hh + r) * D + col] += acc[j][r] + bias;
      }
    }
    __syncthreads();
    layer_norm(xf, xh, P.ln1_g + l * D, P.ln1_b + l * D, wave, lane);
    __syncthreads();

    // ---- FFN1: [64x256] @ [256x1024], relu -> hbuf f16; 64 groups ----
    for (int g = wave; g < 64; g += 8) {
      const int mt = g & 3;
      const int n0 = __builtin_amdgcn_readfirstlane((g >> 2) * 64);
      v8f acc[4] = {{}, {}, {}, {}};
      gemm4<8>(xh, D, mt * 16, W1, D, n0, acc);
#pragma unroll
      for (int j = 0; j < 4; ++j) {
        const int col = n0 + j * 16 + nl;
        const float bias = P.f1_b[l * F + col];
#pragma unroll
        for (int r = 0; r < 8; ++r)
          hbuf[(mt * 16 + 8 * hh + r) * F + col] =
              (_Float16)fmaxf(acc[j][r] + bias, 0.f);
      }
    }
    __syncthreads();

    // ---- FFN2: [64x1024] @ [1024x256] + residual; 16 groups, depth 32 ----
    for (int g = wave; g < 16; g += 8) {
      const int mt = g & 3;
      const int n0 = __builtin_amdgcn_readfirstlane((g >> 2) * 64);
      v8f acc[4] = {{}, {}, {}, {}};
      gemm4<32>(hbuf, F, mt * 16, W2, F, n0, acc);
#pragma unroll
      for (int j = 0; j < 4; ++j) {
        const int col = n0 + j * 16 + nl;
        const float bias = P.f2_b[l * D + col];
#pragma unroll
        for (int r = 0; r < 8; ++r)
          xf[(mt * 16 + 8 * hh + r) * D + col] += acc[j][r] + bias;
      }
    }
    __syncthreads();
    layer_norm(xf, xh, P.ln2_g + l * D, P.ln2_b + l * D, wave, lane);

    // warm next layer's QKV weights in L2 (global_prefetch_b8)
    if (l < 3) {
      const _Float16* nw = P.qkvWt + (size_t)(l + 1) * 768 * D;
      for (int i = tid; i < 1536; i += 256)
        __builtin_prefetch((const void*)(nw + (size_t)i * 128), 0, 1);
    }
    __syncthreads();
  }

  // cls features (row 0, post-LN) -> workspace f16
  P.clsOut[(size_t)b * D + tid] = xh[tid];
}

// ---- head 1: h1 = relu(cls @ v1_W + b), action = tanh(cls @ out_W + b) ----
__launch_bounds__(256)
__global__ void head1_kernel(const _Float16* __restrict__ cls,
                             const _Float16* __restrict__ v1Wt,
                             const float* __restrict__ v1_b,
                             const float* __restrict__ out_W,
                             const float* __restrict__ out_b,
                             _Float16* __restrict__ h1,
                             float* __restrict__ outAct) {
  const int m0 = blockIdx.x * 16;
  const int tid = threadIdx.x, wave = tid >> 5, lane = tid & 31;
  const int nl = lane & 15, hh = lane >> 4;
  for (int nt = wave; nt < 16; nt += 8) {
    v8f acc = {};
#pragma unroll
    for (int kt = 0; kt < 8; ++kt)
      acc = wmma16(loadA(cls, D, m0, kt * 32),
                   loadB(v1Wt, D, nt * 16, kt * 32), acc);
    const int col = nt * 16 + nl;
    const float bias = v1_b[col];
#pragma unroll
    for (int r = 0; r < 8; ++r)
      h1[(size_t)(m0 + 8 * hh + r) * D + col] =
          (_Float16)fmaxf(acc[r] + bias, 0.f);
  }
  if (tid < 32) {
    const int s = tid >> 1, c = tid & 1;
    const _Float16* cr = cls + (size_t)(m0 + s) * D;
    float sum = out_b[c];
    for (int k = 0; k < D; ++k) sum += (float)cr[k] * out_W[k * 2 + c];
    outAct[(size_t)(m0 + s) * 2 + c] = tanhf(sum);
  }
}

// ---- head 2: h2 = relu(h1 @ v2_W + b), value = h2 @ v3_W + b --------------
__launch_bounds__(256)
__global__ void head2_kernel(const _Float16* __restrict__ h1,
                             const _Float16* __restrict__ v2Wt,
                             const float* __restrict__ v2_b,
                             const float* __restrict__ v3_W,
                             const float* __restrict__ v3_b,
                             float* __restrict__ outVal) {
  __shared__ float h2[16 * D];
  const int m0 = blockIdx.x * 16;
  const int tid = threadIdx.x, wave = tid >> 5, lane = tid & 31;
  const int nl = lane & 15, hh = lane >> 4;
  for (int nt = wave; nt < 16; nt += 8) {
    v8f acc = {};
#pragma unroll
    for (int kt = 0; kt < 8; ++kt)
      acc = wmma16(loadA(h1, D, m0, kt * 32),
                   loadB(v2Wt, D, nt * 16, kt * 32), acc);
    const int col = nt * 16 + nl;
    const float bias = v2_b[col];
#pragma unroll
    for (int r = 0; r < 8; ++r)
      h2[(8 * hh + r) * D + col] = fmaxf(acc[r] + bias, 0.f);
  }
  __syncthreads();
  if (tid < 16) {
    float sum = v3_b[0];
    const float* hr = h2 + tid * D;
    for (int k = 0; k < D; ++k) sum += hr[k] * v3_W[k];
    outVal[m0 + tid] = sum;
  }
}

// ---------------------------------------------------------------------------
extern "C" void kernel_launch(void* const* d_in, const int* in_sizes, int n_in,
                              void* d_out, int out_size, void* d_ws, size_t ws_size,
                              hipStream_t stream) {
  const float* obs       = (const float*)d_in[0];
  const float* cls_emb   = (const float*)d_in[1];
  const float* type_cls  = (const float*)d_in[2];
  const float* type_ctx  = (const float*)d_in[3];
  const float* type_pred = (const float*)d_in[4];
  const float* type_boid = (const float*)d_in[5];
  const float* ctx_W     = (const float*)d_in[6];
  const float* ctx_b     = (const float*)d_in[7];
  const float* pred_W    = (const float*)d_in[8];
  const float* pred_b    = (const float*)d_in[9];
  const float* boid_W    = (const float*)d_in[10];
  const float* boid_b    = (const float*)d_in[11];
  const float* qkv_W     = (const float*)d_in[12];
  const float* qkv_b     = (const float*)d_in[13];
  const float* attn_W    = (const float*)d_in[14];
  const float* attn_b    = (const float*)d_in[15];
  const float* ln1_g     = (const float*)d_in[16];
  const float* ln1_b     = (const float*)d_in[17];
  const float* ln2_g     = (const float*)d_in[18];
  const float* ln2_b     = (const float*)d_in[19];
  const float* ffn_W1    = (const float*)d_in[20];
  const float* ffn_b1    = (const float*)d_in[21];
  const float* ffn_W2    = (const float*)d_in[22];
  const float* ffn_b2    = (const float*)d_in[23];
  const float* out_W     = (const float*)d_in[24];
  const float* out_b     = (const float*)d_in[25];
  const float* v1_W      = (const float*)d_in[26];
  const float* v1_b      = (const float*)d_in[27];
  const float* v2_W      = (const float*)d_in[28];
  const float* v2_b      = (const float*)d_in[29];
  const float* v3_W      = (const float*)d_in[30];
  const float* v3_b      = (const float*)d_in[31];

  const int Bn = in_sizes[0] / OBSN;   // 8192

  // workspace layout (f16 elements); ~15 MB total
  _Float16* w = (_Float16*)d_ws;
  _Float16* qkvWt = w;  w += 4 * 768 * D;
  _Float16* attnWt = w; w += 4 * D * D;
  _Float16* f1Wt = w;   w += 4 * F * D;
  _Float16* f2Wt = w;   w += 4 * D * F;
  _Float16* v1Wt = w;   w += D * D;
  _Float16* v2Wt = w;   w += D * D;
  _Float16* clsF = w;   w += (size_t)Bn * D;
  _Float16* h1F  = w;   w += (size_t)Bn * D;
  (void)ws_size; (void)n_in; (void)out_size;

  // ---- weight prep: f32 [K][N] -> f16 [N][K] ----
  auto prep = [&](_Float16* dst, const float* src, int K, int N) {
    int tot = K * N;
    wprep_kernel<<<(tot + 255) / 256, 256, 0, stream>>>(dst, src, K, N);
  };
  for (int l = 0; l < 4; ++l) {
    prep(qkvWt  + (size_t)l * 768 * D, qkv_W  + (size_t)l * D * 768, D, 768);
    prep(attnWt + (size_t)l * D * D,   attn_W + (size_t)l * D * D,   D, D);
    prep(f1Wt   + (size_t)l * F * D,   ffn_W1 + (size_t)l * D * F,   D, F);
    prep(f2Wt   + (size_t)l * D * F,   ffn_W2 + (size_t)l * F * D,   F, D);
  }
  prep(v1Wt, v1_W, D, D);
  prep(v2Wt, v2_W, D, D);

  // ---- main transformer ----
  PolicyParams P;
  P.obs = obs;
  P.cls_emb = cls_emb; P.type_cls = type_cls; P.type_ctx = type_ctx;
  P.type_pred = type_pred; P.type_boid = type_boid;
  P.ctx_W = ctx_W; P.ctx_b = ctx_b;
  P.pred_W = pred_W; P.pred_b = pred_b;
  P.boid_W = boid_W; P.boid_b = boid_b;
  P.qkvWt = qkvWt; P.qkv_b = qkv_b;
  P.attnWt = attnWt; P.attn_b = attn_b;
  P.ln1_g = ln1_g; P.ln1_b = ln1_b; P.ln2_g = ln2_g; P.ln2_b = ln2_b;
  P.f1Wt = f1Wt; P.f1_b = ffn_b1;
  P.f2Wt = f2Wt; P.f2_b = ffn_b2;
  P.clsOut = clsF;

  (void)hipFuncSetAttribute((const void*)policy_kernel,
                            hipFuncAttributeMaxDynamicSharedMemorySize,
                            SMEM_BYTES);
  policy_kernel<<<Bn, 256, SMEM_BYTES, stream>>>(P);

  // ---- heads ----
  float* outAct = (float*)d_out;            // (B,2)
  float* outVal = (float*)d_out + 2 * Bn;   // (B,1)
  head1_kernel<<<Bn / 16, 256, 0, stream>>>(clsF, v1Wt, v1_b, out_W, out_b,
                                            h1F, outAct);
  head2_kernel<<<Bn / 16, 256, 0, stream>>>(h1F, v2Wt, v2_b, v3_W, v3_b,
                                            outVal);
}
